// SorscherRNN_50182397887146
// MI455X (gfx1250) — matrix operand order repaired and checked by reading, not compiled, not run on hardware
//
#include <hip/hip_runtime.h>
#include <cstdint>

#define NG 4096
#define NP 512
#define BB 128
#define TT 100

typedef __attribute__((ext_vector_type(16))) __bf16 v16bf;
typedef __attribute__((ext_vector_type(8)))  float  v8f;

union Frag {
  v16bf v;
  uint4 q[2];
};

__device__ __forceinline__ unsigned short f32_to_bf16(float f) {
  union { float f; unsigned int u; } c; c.f = f;
  unsigned int u = c.u;
  u += 0x7FFFu + ((u >> 16) & 1u);   // round-to-nearest-even
  return (unsigned short)(u >> 16);
}

// -------- one-time: fp32 [R,C] -> bf16 [C,R] transpose (LDS tiled, coalesced) ------
__global__ __launch_bounds__(256) void transpose_f32_bf16(
    const float* __restrict__ src, unsigned short* __restrict__ dst, int R, int C) {
  __shared__ unsigned short tile[32][33];
  const int tx = threadIdx.x;   // 0..31
  const int ty = threadIdx.y;   // 0..7
  const int c0 = blockIdx.x * 32;
  const int r0 = blockIdx.y * 32;
#pragma unroll
  for (int i = 0; i < 32; i += 8)
    tile[ty + i][tx] = f32_to_bf16(src[(size_t)(r0 + ty + i) * C + (c0 + tx)]);
  __syncthreads();
#pragma unroll
  for (int i = 0; i < 32; i += 8)
    dst[(size_t)(c0 + ty + i) * R + (r0 + tx)] = tile[tx][ty + i];
}

// -------- h0 = p0 @ W_enc  (fp32 accumulate, bf16 out; tiny: 0.27 GFLOP) ----------
__global__ __launch_bounds__(256) void h0_kernel(
    const float* __restrict__ p0, const float* __restrict__ Wenc,
    unsigned short* __restrict__ h) {
  const int idx = blockIdx.x * 256 + threadIdx.x;   // 0 .. 128*4096-1
  const int b = idx >> 12;
  const int g = idx & (NG - 1);
  const float* pr = p0 + b * NP;
  float s = 0.f;
#pragma unroll 4
  for (int k = 0; k < NP; k += 4) {
    float4 p = *(const float4*)(pr + k);
    s += p.x * Wenc[(size_t)(k + 0) * NG + g];
    s += p.y * Wenc[(size_t)(k + 1) * NG + g];
    s += p.z * Wenc[(size_t)(k + 2) * NG + g];
    s += p.w * Wenc[(size_t)(k + 3) * NG + g];
  }
  h[idx] = f32_to_bf16(s);
}

// -------- per-step recurrence: h_new = relu(v_t @ W_in + h_prev @ W_rec) ----------
// Wave computes a 16(M) x 32(N) slab; block of 8 waves tiles a 64x64 output block
// (4 M-tiles x 2 N-columns) to minimize L2 traffic: 128 MB/step vs 272 MB for a
// 16x256 block tile. K tiled by 32 with bf16 WMMA, fp32 accumulation.
// hprev: bf16 [128,4096] row-major (A).  WrecT: bf16 [4096(N),4096(K)] (B, K-contig).
__global__ __launch_bounds__(256) void step_kernel(
    const unsigned short* __restrict__ hprev,
    const unsigned short* __restrict__ WrecT,
    const float* __restrict__ v,
    const float* __restrict__ Win,
    unsigned short* __restrict__ hnew,
    int t) {
  const int lane  = threadIdx.x & 31;
  const int wave  = threadIdx.x >> 5;     // 0..7
  const int l15   = lane & 15;
  const int khalf = lane >> 4;            // 0 or 1
  const int mbase = blockIdx.y * 64 + (wave >> 1) * 16;   // 4 M-tiles per block
  const int nbase = blockIdx.x * 64 + (wave & 1) * 32;    // 2 N-columns per block

  // A fragment lane view: row = l15; K offsets {8*khalf+0..7, 8*khalf+16..23}
  const unsigned short* aRow  = hprev + (size_t)(mbase + l15) * NG + 8 * khalf;
  // B fragment lane view: col = l15; K offsets {8*khalf+0..7, 8*khalf+16..23}
  const unsigned short* bRow0 = WrecT + (size_t)(nbase + l15) * NG + 8 * khalf;
  const unsigned short* bRow1 = bRow0 + (size_t)16 * NG;

  v8f acc[2] = {v8f{}, v8f{}};
  for (int k0 = 0; k0 < NG; k0 += 32) {
    Frag a, b0, b1;
    a.q[0]  = *(const uint4*)(aRow + k0);
    a.q[1]  = *(const uint4*)(aRow + k0 + 16);
    b0.q[0] = *(const uint4*)(bRow0 + k0);
    b0.q[1] = *(const uint4*)(bRow0 + k0 + 16);
    b1.q[0] = *(const uint4*)(bRow1 + k0);
    b1.q[1] = *(const uint4*)(bRow1 + k0 + 16);
    acc[0] = __builtin_amdgcn_wmma_f32_16x16x32_bf16(false, a.v, false, b0.v,
                                                     (short)0, acc[0], false, false);
    acc[1] = __builtin_amdgcn_wmma_f32_16x16x32_bf16(false, a.v, false, b1.v,
                                                     (short)0, acc[1], false, false);
  }

  // epilogue: + v_t @ W_in (on the fly), relu, bf16 store
  const float2* v2 = (const float2*)v;
#pragma unroll
  for (int j = 0; j < 2; ++j) {
    const int ncol = nbase + j * 16 + l15;
    const float w0 = Win[ncol];
    const float w1 = Win[NG + ncol];
#pragma unroll
    for (int r = 0; r < 8; ++r) {
      const int m = mbase + r + 8 * khalf;      // C/D layout: VGPR r, lane half
      const float2 vel = v2[m * TT + t];
      float val = acc[j][r] + vel.x * w0 + vel.y * w1;
      val = fmaxf(val, 0.0f);
      hnew[(size_t)m * NG + ncol] = f32_to_bf16(val);
    }
  }
}

// -------- per-step decoder: out[:,t,:] = h_t @ W_dec  (fp32 out) ------------------
__global__ __launch_bounds__(256) void dec_kernel(
    const unsigned short* __restrict__ h,
    const unsigned short* __restrict__ WdecT,   // bf16 [512(N),4096(K)]
    float* __restrict__ out, int t) {
  const int lane  = threadIdx.x & 31;
  const int wave  = threadIdx.x >> 5;
  const int l15   = lane & 15;
  const int khalf = lane >> 4;
  const int mbase = blockIdx.y * 64 + (wave >> 1) * 16;
  const int pbase = blockIdx.x * 64 + (wave & 1) * 32;

  const unsigned short* aRow  = h + (size_t)(mbase + l15) * NG + 8 * khalf;
  const unsigned short* bRow0 = WdecT + (size_t)(pbase + l15) * NG + 8 * khalf;
  const unsigned short* bRow1 = bRow0 + (size_t)16 * NG;

  v8f acc[2] = {v8f{}, v8f{}};
  for (int k0 = 0; k0 < NG; k0 += 32) {
    Frag a, b0, b1;
    a.q[0]  = *(const uint4*)(aRow + k0);
    a.q[1]  = *(const uint4*)(aRow + k0 + 16);
    b0.q[0] = *(const uint4*)(bRow0 + k0);
    b0.q[1] = *(const uint4*)(bRow0 + k0 + 16);
    b1.q[0] = *(const uint4*)(bRow1 + k0);
    b1.q[1] = *(const uint4*)(bRow1 + k0 + 16);
    acc[0] = __builtin_amdgcn_wmma_f32_16x16x32_bf16(false, a.v, false, b0.v,
                                                     (short)0, acc[0], false, false);
    acc[1] = __builtin_amdgcn_wmma_f32_16x16x32_bf16(false, a.v, false, b1.v,
                                                     (short)0, acc[1], false, false);
  }

#pragma unroll
  for (int j = 0; j < 2; ++j) {
    const int pcol = pbase + j * 16 + l15;
#pragma unroll
    for (int r = 0; r < 8; ++r) {
      const int m = mbase + r + 8 * khalf;
      out[(size_t)(m * TT + t) * NP + pcol] = acc[j][r];
    }
  }
}

extern "C" void kernel_launch(void* const* d_in, const int* in_sizes, int n_in,
                              void* d_out, int out_size, void* d_ws, size_t ws_size,
                              hipStream_t stream) {
  (void)in_sizes; (void)n_in; (void)out_size; (void)ws_size;
  const float* v     = (const float*)d_in[0];   // [128,100,2]
  const float* p0    = (const float*)d_in[1];   // [128,512]
  const float* W_enc = (const float*)d_in[2];   // [512,4096]
  const float* W_in  = (const float*)d_in[3];   // [2,4096]
  const float* W_rec = (const float*)d_in[4];   // [4096,4096]
  const float* W_dec = (const float*)d_in[5];   // [4096,512]
  float* out = (float*)d_out;                   // [128,100,512]

  char* ws = (char*)d_ws;
  unsigned short* WrecT = (unsigned short*)ws;                            // 32 MB
  unsigned short* WdecT = (unsigned short*)(ws + (size_t)NG * NG * 2);    //  4 MB
  unsigned short* hA    = (unsigned short*)(ws + (size_t)NG * NG * 2
                                               + (size_t)NP * NG * 2);    //  1 MB
  unsigned short* hB    = hA + (size_t)BB * NG;                           //  1 MB

  // one-time (per call) weight conversion/transpose to bf16 [N,K]
  transpose_f32_bf16<<<dim3(NG / 32, NG / 32), dim3(32, 8), 0, stream>>>(W_rec, WrecT, NG, NG);
  transpose_f32_bf16<<<dim3(NP / 32, NG / 32), dim3(32, 8), 0, stream>>>(W_dec, WdecT, NG, NP);

  // initial hidden state
  h0_kernel<<<(BB * NG) / 256, 256, 0, stream>>>(p0, W_enc, hA);

  unsigned short* hp = hA;
  unsigned short* hn = hB;
  for (int t = 0; t < TT; ++t) {
    // 64x64 block tile: grid (N/64, M/64)
    step_kernel<<<dim3(NG / 64, BB / 64), 256, 0, stream>>>(hp, WrecT, v, W_in, hn, t);
    dec_kernel<<<dim3(NP / 64, BB / 64), 256, 0, stream>>>(hn, WdecT, out, t);
    unsigned short* tmp = hp; hp = hn; hn = tmp;
  }
}
